// ThreeLayerCNN_25099788878322
// MI455X (gfx1250) — compile-verified, matrix operand order checked by reading, not run on hardware
//
#include <hip/hip_runtime.h>
#include <hip/hip_bf16.h>
#include <math.h>

// ---------------------------------------------------------------------------
// Types for CDNA5 WMMA / TDM
// ---------------------------------------------------------------------------
typedef __attribute__((ext_vector_type(16))) _Float16     v16h;
typedef __attribute__((ext_vector_type(8)))  float        v8f;
typedef __attribute__((ext_vector_type(4)))  unsigned int v4u;
typedef __attribute__((ext_vector_type(8)))  int          v8i;
typedef __attribute__((ext_vector_type(4)))  int          v4i;

#define GEMM_BM 64
#define GEMM_BN 32
#define GEMM_KC 32

#if __has_builtin(__builtin_amdgcn_tensor_load_to_lds) && \
    __has_builtin(__builtin_amdgcn_s_wait_tensorcnt)
#define USE_TDM 1
#else
#define USE_TDM 0
#endif

#if USE_TDM
// ---------------------------------------------------------------------------
// Issue one TDM tile load: fp32 tile [tile_rows x tile_cols] from a row-major
// matrix (row stride = stride_elems) into LDS at lds_off, row-major packed.
// rem_cols/rem_rows are the remaining extents measured from the tile origin;
// TDM zero-fills reads beyond them (handles K tails and N edges in hardware).
// D# layout per CDNA5 ISA ch.8 (group0 128b, group1 256b; groups 2/3 unused
// for 2D tiles). This toolchain's builtin takes 6 args (g0,g1,g2,g3,g4,cpol).
// ---------------------------------------------------------------------------
static __device__ __forceinline__ void tdm_load_tile_f32(
    const float* gptr, unsigned lds_off, int rem_cols, int rem_rows,
    int tile_cols, int tile_rows, int stride_elems)
{
    unsigned long long ga = (unsigned long long)(size_t)gptr;
    v4u g0;
    g0[0] = 1u;                                       // count=1, no gather
    g0[1] = lds_off;                                  // lds_addr (bytes)
    g0[2] = (unsigned)(ga & 0xFFFFFFFFu);             // global_addr[31:0]
    g0[3] = (unsigned)((ga >> 32) & 0x01FFFFFFu)      // global_addr[56:32]
            | 0x80000000u;                            // type=2 ("image")
    v8i g1;
    g1[0] = 0x00020000;                               // data_size=2 (4 bytes)
    g1[1] = (int)(((unsigned)rem_cols & 0xFFFFu) << 16);          // tensor_dim0[15:0]
    g1[2] = (int)((((unsigned)rem_cols >> 16) & 0xFFFFu)
            | (((unsigned)rem_rows & 0xFFFFu) << 16));            // dim0 hi | dim1 lo
    g1[3] = (int)((((unsigned)rem_rows >> 16) & 0xFFFFu)
            | (((unsigned)tile_cols & 0xFFFFu) << 16));           // dim1 hi | tile_dim0
    g1[4] = (int)((unsigned)tile_rows & 0xFFFFu);                 // tile_dim1 (tile_dim2=0)
    g1[5] = stride_elems;                                         // tensor_dim0_stride[31:0]
    g1[6] = 0;                                                    // stride hi / dim1_stride
    g1[7] = 0;
    v4i z4 = {0, 0, 0, 0};
    v8i z8 = {0, 0, 0, 0, 0, 0, 0, 0};
    __builtin_amdgcn_tensor_load_to_lds(g0, g1, z4, z4, z8, 0);
}
#endif

// ---------------------------------------------------------------------------
// Generic FC GEMM: C[M,N] = A[M,K] @ W[N,K]^T + bias[N]   (torch Linear)
// M must be a multiple of 64 (we always launch with M=4096).
// fp32 tiles double-buffered in LDS; fed by the Tensor Data Mover (chunk i+1
// issued before computing chunk i, drained with s_wait_tensorcnt<=2 so two
// DMAs stay in flight under the WMMA). Converted to f16 at fragment-pack
// time, WMMA f32 accumulation.
// Block: 256 threads (8 waves). Block tile 64Mx32N; each wave one 16x16 tile.
// ---------------------------------------------------------------------------
__global__ __launch_bounds__(256)
void gemm_f16_wmma(const float* __restrict__ A, const float* __restrict__ W,
                   const float* __restrict__ bias, float* __restrict__ C,
                   int M, int N, int K, int relu)
{
    __shared__ float Asf[2][GEMM_BM * GEMM_KC];  // [m][k], k contiguous (2x8 KB)
    __shared__ float Bsf[2][GEMM_BN * GEMM_KC];  // [n][k], k contiguous (2x4 KB)

    const int tid  = threadIdx.x;
    const int lane = tid & 31;
    const int wave = tid >> 5;              // 0..7
    const int tm   = (wave >> 1) * 16;      // 0,16,32,48
    const int tn   = (wave & 1) * 16;       // 0,16
    const int m0   = blockIdx.x * GEMM_BM;
    const int n0   = blockIdx.y * GEMM_BN;
    const int nch  = (K + GEMM_KC - 1) / GEMM_KC;

#if USE_TDM
    const unsigned As_off0 = (unsigned)(size_t)(void*)&Asf[0][0];
    const unsigned As_off1 = (unsigned)(size_t)(void*)&Asf[1][0];
    const unsigned Bs_off0 = (unsigned)(size_t)(void*)&Bsf[0][0];
    const unsigned Bs_off1 = (unsigned)(size_t)(void*)&Bsf[1][0];

    // prologue: DMA chunk 0 into buffer 0
    if (wave == 0) {
        tdm_load_tile_f32(A + (size_t)m0 * K, As_off0,
                          K, M - m0, GEMM_KC, GEMM_BM, K);
        tdm_load_tile_f32(W + (size_t)n0 * K, Bs_off0,
                          K, N - n0, GEMM_KC, GEMM_BN, K);
    }
#endif

    v8f acc = {};
    int cur = 0;

    for (int ci = 0; ci < nch; ++ci) {
        const int kc = ci * GEMM_KC;
#if USE_TDM
        if (wave == 0) {
            if (ci + 1 < nch) {
                // issue next chunk's DMAs into the other buffer, then drain
                // only the current chunk (two newest stay in flight).
                const int kn = kc + GEMM_KC;
                tdm_load_tile_f32(A + (size_t)m0 * K + kn,
                                  cur ? As_off0 : As_off1,
                                  K - kn, M - m0, GEMM_KC, GEMM_BM, K);
                tdm_load_tile_f32(W + (size_t)n0 * K + kn,
                                  cur ? Bs_off0 : Bs_off1,
                                  K - kn, N - n0, GEMM_KC, GEMM_BN, K);
                __builtin_amdgcn_s_wait_tensorcnt(2);
            } else {
                __builtin_amdgcn_s_wait_tensorcnt(0);
            }
        }
        __syncthreads();
#else
        // ---- branchless staging: A tile 64x32 (8 elems/thread) ----
        #pragma unroll
        for (int i = tid; i < GEMM_BM * GEMM_KC; i += 256) {
            int r  = i >> 5;               // KC == 32
            int k  = i & 31;
            int gk  = kc + k;
            int gkc = gk < K ? gk : (K - 1);
            float v = A[(size_t)(m0 + r) * K + gkc];
            Asf[cur][i] = (gk < K) ? v : 0.0f;
        }
        // ---- branchless staging: B tile 32x32 (rows of W) ----
        #pragma unroll
        for (int i = tid; i < GEMM_BN * GEMM_KC; i += 256) {
            int r  = i >> 5;
            int k  = i & 31;
            int gn  = n0 + r;
            int gk  = kc + k;
            int gnc = gn < N ? gn : (N - 1);
            int gkc = gk < K ? gk : (K - 1);
            float v = W[(size_t)gnc * K + gkc];
            Bsf[cur][i] = (gn < N && gk < K) ? v : 0.0f;
        }
        __syncthreads();
#endif

        // ---- pack fragments per ISA 16-bit A layout (mirror for B) ----
        const int kbase = (lane >> 4) * 8;
        const int ar = tm + (lane & 15);
        const int br = tn + (lane & 15);
        v16h af, bf;
        #pragma unroll
        for (int i = 0; i < 16; ++i) {
            int j = i >> 1, w = i & 1;
            int k = (j < 4) ? (kbase + 2 * j + w)
                            : (16 + kbase + 2 * (j - 4) + w);
            af[i] = (_Float16)Asf[cur][ar * GEMM_KC + k];
            bf[i] = (_Float16)Bsf[cur][br * GEMM_KC + k];
        }
        acc = __builtin_amdgcn_wmma_f32_16x16x32_f16(
            /*neg_a=*/false, af, /*neg_b=*/false, bf,
            /*c_mod=*/(short)0, acc, /*reuse_a=*/false, /*reuse_b=*/false);
        __syncthreads();
        cur ^= 1;
    }

    // ---- store: C layout lane->(n = lane&15, m = 8*(lane>=16) + r) ----
    const int n  = n0 + tn + (lane & 15);
    const int mb = m0 + tm + (lane >> 4) * 8;
    if (n < N) {
        float bv = bias[n];
        #pragma unroll
        for (int r = 0; r < 8; ++r) {
            float v = acc[r] + bv;
            if (relu) v = fmaxf(v, 0.0f);
            C[(size_t)(mb + r) * N + n] = v;
        }
    }
}

// ---------------------------------------------------------------------------
// conv1 (C_in=1) + relu + 2x2 maxpool : x[4096,1,28,28] -> h1[4096,50,13,13]
// ---------------------------------------------------------------------------
__global__ __launch_bounds__(256)
void conv1_relu_pool(const float* __restrict__ x, const float* __restrict__ w,
                     const float* __restrict__ b, float* __restrict__ h1)
{
    const int total = 4096 * 50 * 13 * 13;
    int idx = blockIdx.x * blockDim.x + threadIdx.x;
    if (idx >= total) return;
    int px = idx % 13; int t = idx / 13;
    int py = t % 13;   t /= 13;
    int o  = t % 50;   int bi = t / 50;

    const float* xb = x + (size_t)bi * 28 * 28;
    float wv[9];
    #pragma unroll
    for (int i = 0; i < 9; ++i) wv[i] = w[o * 9 + i];
    const float bias = b[o];

    float best = 0.0f;               // relu() >= 0, so 0 is a valid identity
    #pragma unroll
    for (int dy = 0; dy < 2; ++dy)
    #pragma unroll
    for (int dx = 0; dx < 2; ++dx) {
        int cy = 2 * py + dy, cx = 2 * px + dx;   // conv output coords (<26)
        float s = bias;
        #pragma unroll
        for (int ky = 0; ky < 3; ++ky)
        #pragma unroll
        for (int kx = 0; kx < 3; ++kx)
            s += xb[(cy + ky) * 28 + (cx + kx)] * wv[ky * 3 + kx];
        best = fmaxf(best, fmaxf(s, 0.0f));
    }
    h1[idx] = best;
}

// ---------------------------------------------------------------------------
// Generic 50->50 3x3 conv + relu + 2x2 maxpool.
// in[4096,50,H,W], w[50,50,3,3], out[4096,50,PH,PW]  (conv out (H-2)x(W-2))
// ---------------------------------------------------------------------------
__global__ __launch_bounds__(256)
void convN_relu_pool(const float* __restrict__ in, const float* __restrict__ w,
                     const float* __restrict__ b, float* __restrict__ out,
                     int H, int W, int PH, int PW)
{
    const int total = 4096 * 50 * PH * PW;
    int idx = blockIdx.x * blockDim.x + threadIdx.x;
    if (idx >= total) return;
    int px = idx % PW; int t = idx / PW;
    int py = t % PH;   t /= PH;
    int o  = t % 50;   int bi = t / 50;

    const float bias = b[o];
    float best = 0.0f;
    for (int dy = 0; dy < 2; ++dy)
    for (int dx = 0; dx < 2; ++dx) {
        int cy = 2 * py + dy, cx = 2 * px + dx;
        float s = bias;
        for (int c = 0; c < 50; ++c) {
            const float* ip = in + (((size_t)bi * 50 + c) * H + cy) * W + cx;
            const float* wp = w + ((size_t)o * 50 + c) * 9;
            #pragma unroll
            for (int ky = 0; ky < 3; ++ky)
            #pragma unroll
            for (int kx = 0; kx < 3; ++kx)
                s += ip[ky * W + kx] * wp[ky * 3 + kx];
        }
        best = fmaxf(best, fmaxf(s, 0.0f));
    }
    out[idx] = best;
}

// ---------------------------------------------------------------------------
// Exit routing + log_softmax. exitK = (argmax(eK)==0) <=> e[0] >= e[1].
// ---------------------------------------------------------------------------
__global__ __launch_bounds__(256)
void finalize_select_logsoftmax(const float* __restrict__ e1,
                                const float* __restrict__ e2,
                                const float* __restrict__ o1,
                                const float* __restrict__ o2,
                                const float* __restrict__ o3,
                                float* __restrict__ out)
{
    int s = blockIdx.x * blockDim.x + threadIdx.x;
    if (s >= 4096) return;
    bool x1 = e1[2 * s] >= e1[2 * s + 1];
    bool x2 = e2[2 * s] >= e2[2 * s + 1];
    const float* src = x1 ? (o1 + (size_t)s * 10)
                          : (x2 ? (o2 + (size_t)s * 10) : (o3 + (size_t)s * 10));
    float v[10], mx = -INFINITY;
    #pragma unroll
    for (int i = 0; i < 10; ++i) { v[i] = src[i]; mx = fmaxf(mx, v[i]); }
    float se = 0.0f;
    #pragma unroll
    for (int i = 0; i < 10; ++i) se += expf(v[i] - mx);
    float l = logf(se);
    #pragma unroll
    for (int i = 0; i < 10; ++i) out[(size_t)s * 10 + i] = v[i] - mx - l;
}

// ---------------------------------------------------------------------------
// Host-side launch
// ---------------------------------------------------------------------------
extern "C" void kernel_launch(void* const* d_in, const int* in_sizes, int n_in,
                              void* d_out, int out_size, void* d_ws, size_t ws_size,
                              hipStream_t stream)
{
    const float* x      = (const float*)d_in[0];
    const float* c1_w   = (const float*)d_in[1];
    const float* c1_b   = (const float*)d_in[2];
    const float* c2_w   = (const float*)d_in[3];
    const float* c2_b   = (const float*)d_in[4];
    const float* c3_w   = (const float*)d_in[5];
    const float* c3_b   = (const float*)d_in[6];
    const float* l1c1_w = (const float*)d_in[7];
    const float* l1c1_b = (const float*)d_in[8];
    const float* l2c1_w = (const float*)d_in[9];
    const float* l2c1_b = (const float*)d_in[10];
    const float* l1c2_w = (const float*)d_in[11];
    const float* l1c2_b = (const float*)d_in[12];
    const float* l2c2_w = (const float*)d_in[13];
    const float* l2c2_b = (const float*)d_in[14];
    const float* l1_w   = (const float*)d_in[15];
    const float* l1_b   = (const float*)d_in[16];
    const float* l2_w   = (const float*)d_in[17];
    const float* l2_b   = (const float*)d_in[18];
    const float* d1_w   = (const float*)d_in[19];
    const float* d1_b   = (const float*)d_in[20];
    const float* d2_w   = (const float*)d_in[21];
    const float* d2_b   = (const float*)d_in[22];
    float* out = (float*)d_out;

    // workspace layout (fp32 elements)
    float* ws = (float*)d_ws;
    const size_t N_H1 = (size_t)4096 * 8450;   // 34,611,200
    const size_t N_H2 = (size_t)4096 * 1250;   //  5,120,000
    const size_t N_Z  = (size_t)4096 * 500;
    const size_t N_O  = (size_t)4096 * 10;
    const size_t N_E  = (size_t)4096 * 2;
    float* h1 = ws;                 // f1 view
    float* h2 = h1 + N_H1;          // f2 view
    float* z1 = h2 + N_H2;
    float* z2 = z1 + N_Z;
    float* z3 = z2 + N_Z;
    float* o1 = z3 + N_Z;
    float* o2 = o1 + N_O;
    float* o3 = o2 + N_O;
    float* e1 = o3 + N_O;
    float* e2 = e1 + N_E;
    float* f3 = e2 + N_E;           // 4096*50

    const int M = 4096;
    dim3 blk(256);

    // stage 1: conv1 + relu + pool -> h1 [4096,50,13,13]
    {
        int total = 4096 * 50 * 13 * 13;
        conv1_relu_pool<<<dim3((total + 255) / 256), blk, 0, stream>>>(x, c1_w, c1_b, h1);
    }
    // exit1 logits + branch-1 classifier
    gemm_f16_wmma<<<dim3(M / GEMM_BM, 1), blk, 0, stream>>>(h1, d1_w, d1_b, e1, M, 2, 8450, 0);
    gemm_f16_wmma<<<dim3(M / GEMM_BM, (500 + GEMM_BN - 1) / GEMM_BN), blk, 0, stream>>>(
        h1, l1c1_w, l1c1_b, z1, M, 500, 8450, 0);
    gemm_f16_wmma<<<dim3(M / GEMM_BM, 1), blk, 0, stream>>>(z1, l2c1_w, l2c1_b, o1, M, 10, 500, 0);

    // stage 2: conv2 + relu + pool -> h2 [4096,50,5,5]
    {
        int total = 4096 * 50 * 5 * 5;
        convN_relu_pool<<<dim3((total + 255) / 256), blk, 0, stream>>>(
            h1, c2_w, c2_b, h2, 13, 13, 5, 5);
    }
    gemm_f16_wmma<<<dim3(M / GEMM_BM, 1), blk, 0, stream>>>(h2, d2_w, d2_b, e2, M, 2, 1250, 0);
    gemm_f16_wmma<<<dim3(M / GEMM_BM, (500 + GEMM_BN - 1) / GEMM_BN), blk, 0, stream>>>(
        h2, l1c2_w, l1c2_b, z2, M, 500, 1250, 0);
    gemm_f16_wmma<<<dim3(M / GEMM_BM, 1), blk, 0, stream>>>(z2, l2c2_w, l2c2_b, o2, M, 10, 500, 0);

    // stage 3: conv3 + relu + pool -> f3 [4096,50]
    {
        int total = 4096 * 50 * 1 * 1;
        convN_relu_pool<<<dim3((total + 255) / 256), blk, 0, stream>>>(
            h2, c3_w, c3_b, f3, 5, 5, 1, 1);
    }
    gemm_f16_wmma<<<dim3(M / GEMM_BM, (500 + GEMM_BN - 1) / GEMM_BN), blk, 0, stream>>>(
        f3, l1_w, l1_b, z3, M, 500, 50, 1);   // relu
    gemm_f16_wmma<<<dim3(M / GEMM_BM, 1), blk, 0, stream>>>(z3, l2_w, l2_b, o3, M, 10, 500, 0);

    // routing + log_softmax
    finalize_select_logsoftmax<<<dim3((4096 + 255) / 256), blk, 0, stream>>>(
        e1, e2, o1, o2, o3, out);
}
